// LlamaAttention_heavy_hitter_2010044694828
// MI455X (gfx1250) — matrix operand order repaired and checked by reading, not compiled, not run on hardware
//
#include <hip/hip_runtime.h>
#include <hip/hip_bf16.h>

// ---------------------------------------------------------------------------
// Types for CDNA5 WMMA (wave32): bf16 16x16x32, f32 accumulate
// ---------------------------------------------------------------------------
typedef __bf16 bf16_t;
typedef __attribute__((ext_vector_type(16))) __bf16 v16bf;
typedef __attribute__((ext_vector_type(8)))  __bf16 bf16x8;
typedef __attribute__((ext_vector_type(8)))  float  v8f;

#define WMMA_BF16(a, b, c) \
    __builtin_amdgcn_wmma_f32_16x16x32_bf16(false, (a), false, (b), (short)0, (c), false, false)

static constexpr int S_LEN   = 2048;
static constexpr int HID     = 2048;
static constexpr int NHEADS  = 16;
static constexpr int HDIM    = 128;
static constexpr int BUDGET  = 204;   // int(0.1 * 2048), heavy and recent
static constexpr float SCALE = 0.08838834764831844f; // 1/sqrt(128)
static constexpr float NEGF  = -3.0e38f;

// ---------------------------------------------------------------------------
// CDNA5 async copy global->LDS (CDNA5 ISA).
// LDS destination address per lane = low 32 bits of the generic pointer
// (aperture rule); tracked on ASYNCcnt.
// ---------------------------------------------------------------------------
__device__ __forceinline__ void async_copy_b128(void* lds_dst, const void* gsrc) {
    unsigned l = (unsigned)(uintptr_t)lds_dst;
    asm volatile("global_load_async_to_lds_b128 %0, %1, off"
                 :
                 : "v"(l), "v"(gsrc)
                 : "memory");
}

__device__ __forceinline__ void wait_asynccnt0() {
    asm volatile("s_wait_asynccnt 0x0" ::: "memory");
}

// ---------------------------------------------------------------------------
// Fragment loaders per CDNA5 ISA 7.12.2 (16-bit A 16x32, B 32x16, C/D 16x16)
// A: lane l: row M=(l&15), K = {8h..8h+7, 16+8h..16+8h+7} with h = l>>4
// B: lane l: col N=l&15, K = {8h..8h+7, 16+8h..16+8h+7} with h = l>>4
// C: lane l: col N=l&15, VGPR r holds row M = r + 8*(l>>4)
// ---------------------------------------------------------------------------
__device__ __forceinline__ v16bf load_frag_a(const bf16_t* base, int ld,
                                             int mBase, int kBase, int lane) {
    const int m = mBase + (lane & 15);
    const int h = lane >> 4;
    const bf16_t* p = base + (size_t)m * ld + kBase;
    union { v16bf v; bf16x8 c[2]; } u;
    u.c[0] = *(const bf16x8*)(p + 8 * h);
    u.c[1] = *(const bf16x8*)(p + 16 + 8 * h);
    return u.v;
}

__device__ __forceinline__ v16bf load_frag_b(const bf16_t* base, int ld,
                                             int nBase, int kBase, int lane) {
    const int n  = nBase + (lane & 15);
    const int kb = kBase + 8 * (lane >> 4);
    const bf16_t* p = base + (size_t)n * ld + kb;
    union { v16bf v; bf16x8 c[2]; } u;
    u.c[0] = *(const bf16x8*)(p);
    u.c[1] = *(const bf16x8*)(p + 16);
    return u.v;
}

__device__ __forceinline__ void store_acc_global(float* C, int ldc, int mBase,
                                                 int nBase, int lane, v8f acc) {
    const int n  = nBase + (lane & 15);
    const int mo = mBase + ((lane >> 4) * 8);
#pragma unroll
    for (int r = 0; r < 8; ++r) C[(size_t)(mo + r) * ldc + n] = acc[r];
}

__device__ __forceinline__ void store_acc_lds(float* base, int ld, int nOff,
                                              int lane, v8f acc) {
    const int n  = nOff + (lane & 15);
    const int mo = (lane >> 4) * 8;
#pragma unroll
    for (int r = 0; r < 8; ++r) base[(mo + r) * ld + n] = acc[r];
}

// ---------------------------------------------------------------------------
// Elementwise kernels
// ---------------------------------------------------------------------------
__global__ void f32_to_bf16_kernel(const float* __restrict__ in,
                                   bf16_t* __restrict__ out, int n) {
    int i = blockIdx.x * blockDim.x + threadIdx.x;
    if (i < n) out[i] = (bf16_t)in[i];
}

__global__ void zero_f32_kernel(float* __restrict__ p, int n) {
    int i = blockIdx.x * blockDim.x + threadIdx.x;
    if (i < n) p[i] = 0.0f;
}

// LLaMA RoPE on f32 Q/K, emit bf16. Layout [s][h*HDIM + e].
__global__ void rope_kernel(const float* __restrict__ Qf, const float* __restrict__ Kf,
                            bf16_t* __restrict__ Qb, bf16_t* __restrict__ Kb) {
    const int half = HDIM / 2;
    int idx = blockIdx.x * blockDim.x + threadIdx.x;
    int total = S_LEN * NHEADS * half;
    if (idx >= total) return;
    int i = idx % half;
    int t = idx / half;
    int h = t % NHEADS;
    int s = t / NHEADS;
    float inv = __powf(10000.0f, -(2.0f * (float)i) / (float)HDIM);
    float ang = (float)s * inv;
    float c = __cosf(ang), sn = __sinf(ang);
    size_t base = (size_t)s * HID + (size_t)h * HDIM;
    float q1 = Qf[base + i], q2 = Qf[base + i + half];
    float k1 = Kf[base + i], k2 = Kf[base + i + half];
    Qb[base + i]        = (bf16_t)(q1 * c - q2 * sn);
    Qb[base + i + half] = (bf16_t)(q2 * c + q1 * sn);
    Kb[base + i]        = (bf16_t)(k1 * c - k2 * sn);
    Kb[base + i + half] = (bf16_t)(k2 * c + k1 * sn);
}

// ---------------------------------------------------------------------------
// C[M,N] = A[M,K] @ B[N,K]^T  (nn.Linear). bf16 in, f32 out.
// Block 256 threads = 8 waves; block tile 128x128; wave tile 32x64 (2x4 WMMA).
// B tile (shared by all 8 waves) is staged into LDS with the CDNA5 async DMA
// path, double-buffered so tile kt+1 streams in behind the WMMAs of tile kt.
// A fragments load directly from global (only 2-way shared, clause well).
// ---------------------------------------------------------------------------
__global__ __launch_bounds__(256) void gemm_bf16_wmma(const bf16_t* __restrict__ A,
                                                      const bf16_t* __restrict__ B,
                                                      float* __restrict__ C,
                                                      int M, int N, int K) {
    const int lane = threadIdx.x & 31;
    const int wave = threadIdx.x >> 5;
    const int wr = wave >> 1;           // 0..3
    const int wc = wave & 1;            // 0..1
    const int mBase  = blockIdx.y * 128 + wr * 32;
    const int nBlock = blockIdx.x * 128;

    __shared__ bf16_t sB[2][128][32];   // [buf][n][k]  (16 KB)

    const int tid  = threadIdx.x;
    const int rowB = tid >> 1;          // 0..127  (2 threads per row)
    const int eOff = (tid & 1) * 16;    // element offset within 32-wide row

    // issue async copy of B tile kt into buffer buf (each thread: 2 x b128)
    auto issue_b_tile = [&](int kt, int buf) {
        const bf16_t* g = B + (size_t)(nBlock + rowB) * K + kt * 32 + eOff;
        async_copy_b128(&sB[buf][rowB][eOff],     g);
        async_copy_b128(&sB[buf][rowB][eOff + 8], g + 8);
    };

    v8f acc[2][4] = {};
    const int nkt = K / 32;

    issue_b_tile(0, 0);

    for (int kt = 0; kt < nkt; ++kt) {
        wait_asynccnt0();
        __syncthreads();
        if (kt + 1 < nkt) issue_b_tile(kt + 1, (kt + 1) & 1);

        const int k = kt * 32;
        if (k + 128 < K)
            __builtin_prefetch(A + (size_t)(mBase + (lane & 15)) * K + k + 128, 0, 1);

        v16bf a0 = load_frag_a(A, K, mBase,      k, lane);
        v16bf a1 = load_frag_a(A, K, mBase + 16, k, lane);

        const bf16_t* Bl = &sB[kt & 1][0][0];
        v16bf b[4];
#pragma unroll
        for (int j = 0; j < 4; ++j)
            b[j] = load_frag_b(Bl, 32, wc * 64 + 16 * j, 0, lane);
#pragma unroll
        for (int j = 0; j < 4; ++j) {
            acc[0][j] = WMMA_BF16(a0, b[j], acc[0][j]);
            acc[1][j] = WMMA_BF16(a1, b[j], acc[1][j]);
        }
        __syncthreads();   // all waves done with this buffer before reuse
    }

    const int nBase = nBlock + wc * 64;
#pragma unroll
    for (int i = 0; i < 2; ++i)
#pragma unroll
        for (int j = 0; j < 4; ++j)
            store_acc_global(C, N, mBase + 16 * i, nBase + 16 * j, lane, acc[i][j]);
}

// ---------------------------------------------------------------------------
// Phase 1: per (head, 64-query block): causal softmax stats (flash pass A),
// then accumulate softmax column sums (pass B). Scores stay in LDS.
// 128 threads = 4 waves, wave w owns query rows qBase+16w .. +15.
// ---------------------------------------------------------------------------
__global__ __launch_bounds__(128) void attn_colsum_kernel(const bf16_t* __restrict__ Qb,
                                                          const bf16_t* __restrict__ Kb,
                                                          float* __restrict__ colsum) {
    const int lane = threadIdx.x & 31;
    const int wave = threadIdx.x >> 5;
    const int h = blockIdx.y;
    const int qBase = blockIdx.x * 64;
    const int myQ = qBase + wave * 16;
    const int ld = HID;

    __shared__ float sS[4][16][33];

    const bf16_t* Qh = Qb + (size_t)h * HDIM;
    const bf16_t* Kh = Kb + (size_t)h * HDIM;

    v16bf qf[4];
#pragma unroll
    for (int c = 0; c < 4; ++c) qf[c] = load_frag_a(Qh, ld, myQ, 32 * c, lane);

    const int nkt = blockIdx.x * 2 + 2;      // key tiles through the diagonal block
    const int myRow = myQ + lane;            // valid for lane < 16

    float rowmax = NEGF, rowsum = 0.0f;

    // ---- pass A: row max / sumexp over causal region ----
    for (int kt = 0; kt < nkt; ++kt) {
        const int kb = kt * 32;
        v8f a0 = {}, a1 = {};
#pragma unroll
        for (int c = 0; c < 4; ++c) {
            v16bf b0 = load_frag_b(Kh, ld, kb,      32 * c, lane);
            v16bf b1 = load_frag_b(Kh, ld, kb + 16, 32 * c, lane);
            a0 = WMMA_BF16(qf[c], b0, a0);
            a1 = WMMA_BF16(qf[c], b1, a1);
        }
        store_acc_lds(&sS[wave][0][0], 33, 0,  lane, a0);
        store_acc_lds(&sS[wave][0][0], 33, 16, lane, a1);
        __syncthreads();
        if (lane < 16) {
            float tmax = NEGF;
#pragma unroll
            for (int c = 0; c < 32; ++c) {
                int key = kb + c;
                if (key <= myRow) tmax = fmaxf(tmax, sS[wave][lane][c] * SCALE);
            }
            if (tmax > -2.0e38f) {
                float nm = fmaxf(rowmax, tmax);
                float corr = (rowmax > -2.0e38f) ? __expf(rowmax - nm) : 0.0f;
                float part = 0.0f;
#pragma unroll
                for (int c = 0; c < 32; ++c) {
                    int key = kb + c;
                    if (key <= myRow) part += __expf(sS[wave][lane][c] * SCALE - nm);
                }
                rowsum = rowsum * corr + part;
                rowmax = nm;
            }
        }
        __syncthreads();
    }

    const float invsum = (rowsum > 0.0f) ? 1.0f / rowsum : 0.0f;

    // ---- pass B: accumulate column sums of softmax probs ----
    for (int kt = 0; kt < nkt; ++kt) {
        const int kb = kt * 32;
        v8f a0 = {}, a1 = {};
#pragma unroll
        for (int c = 0; c < 4; ++c) {
            v16bf b0 = load_frag_b(Kh, ld, kb,      32 * c, lane);
            v16bf b1 = load_frag_b(Kh, ld, kb + 16, 32 * c, lane);
            a0 = WMMA_BF16(qf[c], b0, a0);
            a1 = WMMA_BF16(qf[c], b1, a1);
        }
        store_acc_lds(&sS[wave][0][0], 33, 0,  lane, a0);
        store_acc_lds(&sS[wave][0][0], 33, 16, lane, a1);
        __syncthreads();
        if (lane < 16) {
#pragma unroll
            for (int c = 0; c < 32; ++c) {
                int key = kb + c;
                float p = 0.0f;
                if (key <= myRow)
                    p = __expf(sS[wave][lane][c] * SCALE - rowmax) * invsum;
                sS[wave][lane][c] = p;
            }
        }
        __syncthreads();
        if (threadIdx.x < 32) {
            float part = 0.0f;
#pragma unroll
            for (int w = 0; w < 4; ++w)
                for (int r = 0; r < 16; ++r) part += sS[w][r][threadIdx.x];
            atomicAdd(&colsum[(size_t)h * S_LEN + kb + threadIdx.x], part);
        }
        __syncthreads();
    }
}

// ---------------------------------------------------------------------------
// Per-head top-k (204 of 2048) -> heavy mask
// ---------------------------------------------------------------------------
__global__ __launch_bounds__(256) void topk_kernel(const float* __restrict__ colsum,
                                                   int* __restrict__ heavy) {
    const int h = blockIdx.x;
    const int tid = threadIdx.x;
    __shared__ float vals[S_LEN];
    __shared__ float rv[256];
    __shared__ int   ri[256];

    for (int i = tid; i < S_LEN; i += blockDim.x) {
        vals[i] = colsum[(size_t)h * S_LEN + i];
        heavy[(size_t)h * S_LEN + i] = 0;
    }
    __syncthreads();

    for (int it = 0; it < BUDGET; ++it) {
        float bv = NEGF; int bi = S_LEN;
        for (int i = tid; i < S_LEN; i += blockDim.x) {
            float v = vals[i];
            if (v > bv || (v == bv && i < bi)) { bv = v; bi = i; }
        }
        rv[tid] = bv; ri[tid] = bi;
        __syncthreads();
        for (int off = 128; off > 0; off >>= 1) {
            if (tid < off) {
                if (rv[tid + off] > rv[tid] ||
                    (rv[tid + off] == rv[tid] && ri[tid + off] < ri[tid])) {
                    rv[tid] = rv[tid + off];
                    ri[tid] = ri[tid + off];
                }
            }
            __syncthreads();
        }
        if (tid == 0) {
            int b = ri[0];
            heavy[(size_t)h * S_LEN + b] = 1;
            vals[b] = NEGF;
        }
        __syncthreads();
    }
}

// ---------------------------------------------------------------------------
// Phase 2: masked flash attention (causal && (heavy || recent band)) + P @ V.
// 128 threads = 4 waves; wave = 16 query rows, full d=128 accumulated (8 v8f).
// ---------------------------------------------------------------------------
__global__ __launch_bounds__(128) void attn_out_kernel(const bf16_t* __restrict__ Qb,
                                                       const bf16_t* __restrict__ Kb,
                                                       const bf16_t* __restrict__ Vb,
                                                       const int* __restrict__ heavy,
                                                       bf16_t* __restrict__ attnb) {
    const int lane = threadIdx.x & 31;
    const int wave = threadIdx.x >> 5;
    const int h = blockIdx.y;
    const int qBase = blockIdx.x * 64;
    const int myQ = qBase + wave * 16;
    const int ld = HID;

    __shared__ float  sS[4][16][33];
    __shared__ bf16_t sP[4][16][32];
    __shared__ bf16_t sVt[HDIM][32];     // V tile transposed: [d][key]
    __shared__ float  sScale[4][16];
    __shared__ float  sSum[4][16];

    const bf16_t* Qh = Qb + (size_t)h * HDIM;
    const bf16_t* Kh = Kb + (size_t)h * HDIM;
    const bf16_t* Vh = Vb + (size_t)h * HDIM;
    const int* hv = heavy + (size_t)h * S_LEN;

    v16bf qf[4];
#pragma unroll
    for (int c = 0; c < 4; ++c) qf[c] = load_frag_a(Qh, ld, myQ, 32 * c, lane);

    v8f oacc[8] = {};
    float rowmax = NEGF, rowsum = 0.0f;
    const int myRow = myQ + lane;        // valid for lane < 16
    const int nkt = blockIdx.x * 2 + 2;

    for (int kt = 0; kt < nkt; ++kt) {
        const int kb = kt * 32;

        // stage V tile transposed into LDS (coalesced row reads)
        {
            const int k = threadIdx.x >> 2;            // 0..31
            const int dseg = (threadIdx.x & 3) * 32;   // 0,32,64,96
            const bf16_t* src = Vh + (size_t)(kb + k) * ld + dseg;
#pragma unroll
            for (int d = 0; d < 32; ++d) sVt[dseg + d][k] = src[d];
        }

        // scores
        v8f a0 = {}, a1 = {};
#pragma unroll
        for (int c = 0; c < 4; ++c) {
            v16bf b0 = load_frag_b(Kh, ld, kb,      32 * c, lane);
            v16bf b1 = load_frag_b(Kh, ld, kb + 16, 32 * c, lane);
            a0 = WMMA_BF16(qf[c], b0, a0);
            a1 = WMMA_BF16(qf[c], b1, a1);
        }
        store_acc_lds(&sS[wave][0][0], 33, 0,  lane, a0);
        store_acc_lds(&sS[wave][0][0], 33, 16, lane, a1);
        __syncthreads();

        // masked online softmax (one lane per row)
        if (lane < 16) {
            float sv[32];
            float tmax = NEGF;
#pragma unroll
            for (int c = 0; c < 32; ++c) {
                int key = kb + c;
                bool valid = (key <= myRow) && (hv[key] || (myRow - key) <= BUDGET);
                sv[c] = valid ? sS[wave][lane][c] * SCALE : NEGF;
                tmax = fmaxf(tmax, sv[c]);
            }
            float nm = fmaxf(rowmax, tmax);
            float corr = (rowmax > -2.0e38f && nm > -2.0e38f) ? __expf(rowmax - nm) : 1.0f;
            float part = 0.0f;
#pragma unroll
            for (int c = 0; c < 32; ++c) {
                float p = (sv[c] > -2.0e38f) ? __expf(sv[c] - nm) : 0.0f;
                sP[wave][lane][c] = (bf16_t)p;
                part += p;
            }
            rowsum = rowsum * corr + part;
            rowmax = nm;
            sScale[wave][lane] = corr;
        }
        __syncthreads();

        // rescale accumulated output, then accumulate P @ V
        {
            const int half = lane >> 4;
            float cr[8];
#pragma unroll
            for (int r = 0; r < 8; ++r) cr[r] = sScale[wave][r + half * 8];
#pragma unroll
            for (int j = 0; j < 8; ++j)
#pragma unroll
                for (int r = 0; r < 8; ++r) oacc[j][r] *= cr[r];
        }
        v16bf ap = load_frag_a(&sP[wave][0][0], 32, 0, 0, lane);
#pragma unroll
        for (int j = 0; j < 8; ++j) {
            v16bf bv = load_frag_b(&sVt[0][0], 32, 16 * j, 0, lane);
            oacc[j] = WMMA_BF16(ap, bv, oacc[j]);
        }
        __syncthreads();   // protect sS/sP/sVt before next iteration
    }

    if (lane < 16) sSum[wave][lane] = (rowsum > 0.0f) ? 1.0f / rowsum : 0.0f;
    __syncthreads();

    {
        const int half = lane >> 4;
        const int n = lane & 15;
        float inv[8];
#pragma unroll
        for (int r = 0; r < 8; ++r) inv[r] = sSum[wave][r + half * 8];
#pragma unroll
        for (int j = 0; j < 8; ++j)
#pragma unroll
            for (int r = 0; r < 8; ++r) {
                int m = myQ + r + half * 8;
                attnb[(size_t)m * ld + (size_t)h * HDIM + 16 * j + n] =
                    (bf16_t)(oacc[j][r] * inv[r]);
            }
    }
}

// ---------------------------------------------------------------------------
// Host launch
// ---------------------------------------------------------------------------
extern "C" void kernel_launch(void* const* d_in, const int* in_sizes, int n_in,
                              void* d_out, int out_size, void* d_ws, size_t ws_size,
                              hipStream_t stream) {
    (void)in_sizes; (void)n_in; (void)out_size; (void)ws_size;

    const float* X  = (const float*)d_in[0];
    const float* Wq = (const float*)d_in[1];
    const float* Wk = (const float*)d_in[2];
    const float* Wv = (const float*)d_in[3];
    const float* Wo = (const float*)d_in[4];

    const size_t E = (size_t)S_LEN * HID;   // 4M elements

    char* ws = (char*)d_ws;
    bf16_t* Xb  = (bf16_t*)ws; ws += E * 2;
    bf16_t* Wqb = (bf16_t*)ws; ws += E * 2;
    bf16_t* Wkb = (bf16_t*)ws; ws += E * 2;
    bf16_t* Wvb = (bf16_t*)ws; ws += E * 2;
    bf16_t* Wob = (bf16_t*)ws; ws += E * 2;
    float*  Qf  = (float*)ws;  ws += E * 4;
    float*  Kf  = (float*)ws;  ws += E * 4;
    float*  Vf  = (float*)ws;  ws += E * 4;
    bf16_t* Qb  = (bf16_t*)ws; ws += E * 2;
    bf16_t* Kb  = (bf16_t*)ws; ws += E * 2;
    bf16_t* Vb  = (bf16_t*)ws; ws += E * 2;
    bf16_t* Ab  = (bf16_t*)ws; ws += E * 2;        // attention output, bf16
    float*  colsum = (float*)ws; ws += (size_t)NHEADS * S_LEN * 4;
    int*    heavy  = (int*)ws;   ws += (size_t)NHEADS * S_LEN * 4;

    const int cb = (int)((E + 255) / 256);
    f32_to_bf16_kernel<<<cb, 256, 0, stream>>>(X,  Xb,  (int)E);
    f32_to_bf16_kernel<<<cb, 256, 0, stream>>>(Wq, Wqb, (int)E);
    f32_to_bf16_kernel<<<cb, 256, 0, stream>>>(Wk, Wkb, (int)E);
    f32_to_bf16_kernel<<<cb, 256, 0, stream>>>(Wv, Wvb, (int)E);
    f32_to_bf16_kernel<<<cb, 256, 0, stream>>>(Wo, Wob, (int)E);

    dim3 gg(HID / 128, S_LEN / 128);
    gemm_bf16_wmma<<<gg, 256, 0, stream>>>(Xb, Wqb, Qf, S_LEN, HID, HID);
    gemm_bf16_wmma<<<gg, 256, 0, stream>>>(Xb, Wkb, Kf, S_LEN, HID, HID);
    gemm_bf16_wmma<<<gg, 256, 0, stream>>>(Xb, Wvb, Vf, S_LEN, HID, HID);

    const int rtot = S_LEN * NHEADS * (HDIM / 2);
    rope_kernel<<<(rtot + 255) / 256, 256, 0, stream>>>(Qf, Kf, Qb, Kb);
    f32_to_bf16_kernel<<<cb, 256, 0, stream>>>(Vf, Vb, (int)E);

    zero_f32_kernel<<<(NHEADS * S_LEN + 255) / 256, 256, 0, stream>>>(colsum, NHEADS * S_LEN);

    dim3 ga(S_LEN / 64, NHEADS);
    attn_colsum_kernel<<<ga, 128, 0, stream>>>(Qb, Kb, colsum);
    topk_kernel<<<NHEADS, 256, 0, stream>>>(colsum, heavy);
    attn_out_kernel<<<ga, 128, 0, stream>>>(Qb, Kb, Vb, heavy, Ab);

    gemm_bf16_wmma<<<gg, 256, 0, stream>>>(Ab, Wob, (float*)d_out, S_LEN, HID, HID);
}
